// GFM_79207786873390
// MI455X (gfx1250) — matrix operand order repaired and checked by reading, not compiled, not run on hardware
//
#include <hip/hip_runtime.h>
#include <hip/hip_bf16.h>

typedef __bf16 bf16;
typedef __attribute__((ext_vector_type(16))) __bf16 v16bf;
typedef __attribute__((ext_vector_type(8)))  __bf16 v8bf;
typedef __attribute__((ext_vector_type(8)))  float  v8f;
typedef __attribute__((ext_vector_type(4)))  unsigned int v4u;
typedef __attribute__((ext_vector_type(8)))  int v8i;
typedef __attribute__((ext_vector_type(4)))  int v4i;

#define DIM   192
#define HW    16384          // 128*128
#define NTOT  131072         // B(8) * HW
#define NB    8
#define HEADS 4
#define CH    48

static __device__ __forceinline__ v16bf mk16(v8bf lo, v8bf hi) {
  return __builtin_shufflevector(lo, hi, 0, 1, 2, 3, 4, 5, 6, 7,
                                 8, 9, 10, 11, 12, 13, 14, 15);
}
// A fragment: two 16B runs at +0 and +16 elements (16B aligned)
static __device__ __forceinline__ v16bf load_a16(const bf16* p) {
  return mk16(*(const v8bf*)p, *(const v8bf*)(p + 16));
}
// B fragment: two 16B runs at +0 and +16 elements (16B aligned)
static __device__ __forceinline__ v16bf load_b16(const bf16* p) {
  return mk16(*(const v8bf*)p, *(const v8bf*)(p + 16));
}

// Tensor-Data-Mover load of a [rows x 192] bf16 row-major tile into LDS.
// D# built per CDNA5 ISA ch.8 (group0: count/lds/global/type, group1: dims).
static __device__ __forceinline__ void tdm_load_tile_192(const bf16* gsrc,
                                                         void* lds_dst, int rows) {
  unsigned long long ga = (unsigned long long)(uintptr_t)gsrc;
  unsigned lds = (unsigned)(uintptr_t)lds_dst;   // LDS aperture: low 32 bits = offset
  v4u g0;
  g0[0] = 1u;                                    // count=1, user descriptor
  g0[1] = lds;                                   // lds_addr
  g0[2] = (unsigned)(ga & 0xFFFFFFFFu);          // global_addr[31:0]
  g0[3] = (unsigned)((ga >> 32) & 0x01FFFFFFu) | (2u << 30);  // addr[56:32] | type=2
  v8i g1;
  g1[0] = (int)(1u << 16);                       // data_size=1 -> 2 bytes
  g1[1] = (int)(192u << 16);                     // tensor_dim0 lo16 (=192)
  g1[2] = (int)((unsigned)rows << 16);           // tensor_dim1 lo16
  g1[3] = (int)(192u << 16);                     // tile_dim0 (=192)
  g1[4] = rows;                                  // tile_dim1
  g1[5] = 192;                                   // tensor_dim0_stride lo32
  g1[6] = 0;
  g1[7] = 0;
  v4i gz = {0, 0, 0, 0};
#if defined(__clang_major__) && (__clang_major__ >= 23)
  v8i gz8 = {0, 0, 0, 0, 0, 0, 0, 0};
  __builtin_amdgcn_tensor_load_to_lds(g0, g1, gz, gz, gz8, 0);
#else
  __builtin_amdgcn_tensor_load_to_lds(g0, g1, gz, gz, 0);
#endif
}

// ---------------------------------------------------------------------------
// fp32 -> bf16 convert (weights)
// ---------------------------------------------------------------------------
__global__ __launch_bounds__(256) void cvt_bf16_kernel(const float* __restrict__ src,
                                                       bf16* __restrict__ dst, int n) {
  int i = blockIdx.x * 256 + threadIdx.x;
  if (i < n) dst[i] = (bf16)src[i];
}

// ---------------------------------------------------------------------------
// LayerNorm2d over channels -> PIXEL-major bf16 [NTOT, 192]
// ---------------------------------------------------------------------------
__global__ __launch_bounds__(256) void ln_kernel(const float* __restrict__ x,
                                                 const float* __restrict__ w,
                                                 const float* __restrict__ bsc,
                                                 bf16* __restrict__ xt) {
  int p  = blockIdx.x * 256 + threadIdx.x;     // 0..NTOT-1
  int b  = p >> 14;
  int hw = p & (HW - 1);
  const float* px = x + (size_t)b * DIM * HW + hw;
  float s = 0.f, s2 = 0.f;
  for (int c = 0; c < DIM; ++c) {
    float v = px[(size_t)c * HW];
    s += v; s2 += v * v;
  }
  float mu   = s * (1.f / DIM);
  float var  = s2 * (1.f / DIM) - mu * mu;
  float rstd = rsqrtf(var + 1e-6f);
  for (int c0 = 0; c0 < DIM; c0 += 8) {
    if (c0 + 8 < DIM) __builtin_prefetch(px + (size_t)(c0 + 8) * HW, 0, 3);
    v8bf o;
#pragma unroll
    for (int j = 0; j < 8; ++j) {
      int c = c0 + j;
      o[j] = (bf16)((px[(size_t)c * HW] - mu) * rstd * w[c] + bsc[c]);
    }
    *(v8bf*)&xt[(size_t)p * DIM + c0] = o;
  }
}

// ---------------------------------------------------------------------------
// GEMM: Y[192, NTOT] = W[192,192]bf16 * X[NTOT,192]bf16 (+bias)(+residual NCHW)
// X is PIXEL-major so B fragments are K-contiguous with zero transposition.
// A tile pulled by the Tensor Data Mover; B tile vector-copied.
// block: 64x64 tile, 256 threads = 8 waves, each wave 16x32 (2 WMMA tiles)
// ---------------------------------------------------------------------------
__global__ __launch_bounds__(256) void gemm192_kernel(const bf16* __restrict__ W,
                                                      const bf16* __restrict__ X,
                                                      const float* __restrict__ bias,
                                                      const float* __restrict__ resid,
                                                      float* __restrict__ Y) {
  __shared__ __align__(16) bf16 As[64][192];    // 64 rows x K(192), TDM-filled
  __shared__ __align__(16) bf16 Bst[64][200];   // 64 pixels x K(192) + pad
  const int tid = threadIdx.x;
  const int m0  = blockIdx.y * 64;
  const int n0  = blockIdx.x * 64;

  // A tile: async TDM DMA (issued once, by wave 0; overlaps B copy below)
  if (tid < 32) tdm_load_tile_192(&W[(size_t)m0 * DIM], (void*)&As[0][0], 64);

  // B tile: straight vector copy (rows = pixels, K contiguous)
  for (int i = tid; i < 64 * 24; i += 256) {
    int r = i / 24, c8 = (i - r * 24) * 8;
    *(v8bf*)&Bst[r][c8] = *(const v8bf*)&X[(size_t)(n0 + r) * DIM + c8];
  }
  if (tid < 32) __builtin_amdgcn_s_wait_tensorcnt((short)0);
  __syncthreads();

  const int wave  = tid >> 5;
  const int lane  = tid & 31;
  const int wm    = wave & 3;              // 4 row groups of 16
  const int wn    = wave >> 2;             // 2 col groups of 32
  const int lrow  = lane & 15;
  const int hi    = lane >> 4;
  const int ka    = hi << 3;               // A: lane-half K offset (0/8)
  const int kbk   = hi << 3;               // B: lane-half K offset (0/8)

  v8f acc0 = {}; v8f acc1 = {};
#pragma unroll
  for (int kb = 0; kb < DIM; kb += 32) {
    v16bf a  = load_a16(&As[wm * 16 + lrow][kb + ka]);
    v16bf b0 = load_b16(&Bst[wn * 32 + lrow][kb + kbk]);
    v16bf b1 = load_b16(&Bst[wn * 32 + 16 + lrow][kb + kbk]);
    acc0 = __builtin_amdgcn_wmma_f32_16x16x32_bf16(false, a, false, b0, (short)0, acc0, false, false);
    acc1 = __builtin_amdgcn_wmma_f32_16x16x32_bf16(false, a, false, b1, (short)0, acc1, false, false);
  }

  const int hi8 = hi << 3;
  if (resid) {
#pragma unroll
    for (int r = 0; r < 8; ++r) {
      int m = m0 + wm * 16 + r + hi8;
      float bv = bias[m];
#pragma unroll
      for (int t = 0; t < 2; ++t) {
        int n = n0 + wn * 32 + t * 16 + lrow;
        int bb = n >> 14, hw = n & (HW - 1);
        float v = (t ? acc1[r] : acc0[r]) + bv +
                  resid[((size_t)bb * DIM + m) * HW + hw];
        Y[(size_t)m * NTOT + n] = v;
      }
    }
  } else {
#pragma unroll
    for (int r = 0; r < 8; ++r) {
      int m = m0 + wm * 16 + r + hi8;
      float bv = bias[m];
#pragma unroll
      for (int t = 0; t < 2; ++t) {
        int n = n0 + wn * 32 + t * 16 + lrow;
        Y[(size_t)m * NTOT + n] = (t ? acc1[r] : acc0[r]) + bv;
      }
    }
  }
}

// ---------------------------------------------------------------------------
// depthwise 3x3 ('SAME', zero pad) per (b,c) plane; bf16 out + optional sum(x^2)
// output stays TOKEN-major [192, NTOT] (what the attention GEMMs want)
// ---------------------------------------------------------------------------
__global__ __launch_bounds__(256) void dwconv_kernel(const float* __restrict__ Yin,
                                                     const float* __restrict__ dwW,
                                                     const float* __restrict__ dwB,
                                                     bf16* __restrict__ Yout,
                                                     float* __restrict__ sumsq) {
  const int c = blockIdx.y;        // 192
  const int b = blockIdx.x;        // 8
  const size_t base = (size_t)c * NTOT + (size_t)b * HW;
  float w[9];
#pragma unroll
  for (int i = 0; i < 9; ++i) w[i] = dwW[c * 9 + i];
  const float bias = dwB[c];
  const int tid = threadIdx.x;
  float ss = 0.f;
  for (int t = 0; t < 64; ++t) {
    int pix = tid + t * 256;
    int y = pix >> 7, x = pix & 127;
    float acc = bias;
#pragma unroll
    for (int dy = -1; dy <= 1; ++dy) {
      int yy = y + dy;
      if (yy < 0 || yy > 127) continue;
#pragma unroll
      for (int dx = -1; dx <= 1; ++dx) {
        int xx = x + dx;
        if (xx < 0 || xx > 127) continue;
        acc += w[(dy + 1) * 3 + (dx + 1)] * Yin[base + yy * 128 + xx];
      }
    }
    Yout[base + pix] = (bf16)acc;
    ss += acc * acc;
  }
  if (sumsq) {
    __shared__ float red[256];
    red[tid] = ss;
    __syncthreads();
    for (int s = 128; s > 0; s >>= 1) {
      if (tid < s) red[tid] += red[tid + s];
      __syncthreads();
    }
    if (tid == 0) sumsq[b * DIM + c] = red[0];
  }
}

// in-place L2 row rescale of bf16 plane
__global__ __launch_bounds__(256) void norm_rows_kernel(bf16* __restrict__ X,
                                                        const float* __restrict__ sumsq) {
  const int c = blockIdx.y, b = blockIdx.x;
  float n  = sqrtf(sumsq[b * DIM + c]);
  float sc = 1.f / fmaxf(n, 1e-12f);
  const size_t base = (size_t)c * NTOT + (size_t)b * HW;
  for (int t = threadIdx.x * 8; t < HW; t += 256 * 8) {
    v8bf v = *(const v8bf*)&X[base + t];
#pragma unroll
    for (int j = 0; j < 8; ++j) v[j] = (bf16)((float)v[j] * sc);
    *(v8bf*)&X[base + t] = v;
  }
}

// ---------------------------------------------------------------------------
// attn[b,h,48,48] = (q_hat k_hat^T) * temperature ; K = 16384 split over 8 waves
// direct vectorized global fragment loads (rows are K-major)
// ---------------------------------------------------------------------------
__global__ __launch_bounds__(256) void attn_kernel(const bf16* __restrict__ Q,
                                                   const bf16* __restrict__ K,
                                                   const float* __restrict__ temperature,
                                                   float* __restrict__ attn) {
  __shared__ float sA[48 * 48];
  const int tid = threadIdx.x;
  const int bh  = blockIdx.x;           // b*4 + h
  const int b   = bh >> 2, h = bh & 3;
  const size_t base = (size_t)(h * CH) * NTOT + (size_t)b * HW;
  for (int i = tid; i < 48 * 48; i += 256) sA[i] = 0.f;
  __syncthreads();

  const int wave = tid >> 5, lane = tid & 31;
  const int lrow = lane & 15;
  const int hi   = lane >> 4;
  const int ka   = hi << 3;
  const int kbk  = hi << 3;

  v8f acc[3][3];
#pragma unroll
  for (int i = 0; i < 3; ++i)
#pragma unroll
    for (int j = 0; j < 3; ++j) acc[i][j] = (v8f){};

  const int k0 = wave * 2048;
  for (int kb = k0; kb < k0 + 2048; kb += 32) {
    v16bf bf[3];
#pragma unroll
    for (int nt = 0; nt < 3; ++nt)
      bf[nt] = load_b16(K + base + (size_t)(nt * 16 + lrow) * NTOT + kb + kbk);
#pragma unroll
    for (int mt = 0; mt < 3; ++mt) {
      v16bf a = load_a16(Q + base + (size_t)(mt * 16 + lrow) * NTOT + kb + ka);
#pragma unroll
      for (int nt = 0; nt < 3; ++nt)
        acc[mt][nt] = __builtin_amdgcn_wmma_f32_16x16x32_bf16(false, a, false, bf[nt],
                                                              (short)0, acc[mt][nt], false, false);
    }
  }
  const int hi8 = hi << 3;
#pragma unroll
  for (int mt = 0; mt < 3; ++mt)
#pragma unroll
    for (int nt = 0; nt < 3; ++nt)
#pragma unroll
      for (int r = 0; r < 8; ++r)
        atomicAdd(&sA[(mt * 16 + r + hi8) * 48 + nt * 16 + lrow], acc[mt][nt][r]);
  __syncthreads();
  float t = temperature[0];
  for (int i = tid; i < 48 * 48; i += 256)
    attn[(size_t)bh * 2304 + i] = sA[i] * t;
}

// ---------------------------------------------------------------------------
// fused 4x top-k softmax + alpha-combine: P = sum_i a_i * softmax_topk_i(attn)
// ---------------------------------------------------------------------------
__global__ __launch_bounds__(64) void topk_kernel(const float* __restrict__ attn,
                                                  const float* __restrict__ pa1,
                                                  const float* __restrict__ pa2,
                                                  const float* __restrict__ pa3,
                                                  const float* __restrict__ pa4,
                                                  bf16* __restrict__ P) {
  __shared__ float tile[48 * 48];
  const int bh = blockIdx.x, tid = threadIdx.x;
  for (int i = tid; i < 2304; i += 64) tile[i] = attn[(size_t)bh * 2304 + i];
  __syncthreads();
  if (tid >= 48) return;
  const float* x = &tile[tid * 48];
  const int ks[4] = {CH / 2, CH * 2 / 3, (CH * 3) / 4, (CH * 4) / 5}; // 24,32,36,38
  const float aw[4] = {pa1[0], pa2[0], pa3[0], pa4[0]};
  float mx = -3.4e38f;
  for (int d = 0; d < 48; ++d) mx = fmaxf(mx, x[d]);
  float sums[4] = {0.f, 0.f, 0.f, 0.f};
  for (int d = 0; d < 48; ++d) {
    int rk = 0;
    for (int d2 = 0; d2 < 48; ++d2) rk += (x[d2] > x[d]) ? 1 : 0;
    float ex = __expf(x[d] - mx);
#pragma unroll
    for (int i = 0; i < 4; ++i)
      if (rk < ks[i]) sums[i] += ex;
  }
  for (int d = 0; d < 48; ++d) {
    int rk = 0;
    for (int d2 = 0; d2 < 48; ++d2) rk += (x[d2] > x[d]) ? 1 : 0;
    float ex = __expf(x[d] - mx);
    float p = 0.f;
#pragma unroll
    for (int i = 0; i < 4; ++i)
      if (rk < ks[i]) p += aw[i] * ex / sums[i];
    P[(size_t)bh * 2304 + tid * 48 + d] = (bf16)p;
  }
}

// ---------------------------------------------------------------------------
// O = P @ V per (b,h); K=48 zero-padded to 64; output PIXEL-major [NTOT, 192]
// block: 48 x 128 tile; 8 waves -> one 16-col strip each, 3 row tiles
// result staged via LDS so global stores are 16B-wide
// ---------------------------------------------------------------------------
__global__ __launch_bounds__(256) void pv_kernel(const bf16* __restrict__ P,
                                                 const bf16* __restrict__ V,
                                                 bf16* __restrict__ O) {
  __shared__ __align__(16) bf16 As[48][72];     // M x Kpad(64)
  __shared__ __align__(16) bf16 Bst[128][72];   // Ncol(128) x Kpad(64)
  __shared__ __align__(16) bf16 Ost[128][56];   // pixel x head-channel staging
  const int bh = blockIdx.y;
  const int b = bh >> 2, h = bh & 3;
  const int n0 = blockIdx.x * 128;
  const int tid = threadIdx.x;

  // P tile (48x48 -> Kpad 64, zero tail), 16B chunks
  for (int i = tid; i < 48 * 8; i += 256) {
    int r = i >> 3, c8 = (i & 7) * 8;
    v8bf v;
    if (c8 < 48) v = *(const v8bf*)&P[(size_t)bh * 2304 + r * 48 + c8];
    else {
#pragma unroll
      for (int j = 0; j < 8; ++j) v[j] = (bf16)0.f;
    }
    *(v8bf*)&As[r][c8] = v;
  }
  // V tile: 8-wide coalesced reads along N, scatter-transpose; zero-pad k>=48
  const size_t vbase = (size_t)(h * CH) * NTOT + (size_t)b * HW + n0;
  for (int i = tid; i < 64 * 16; i += 256) {
    int k = i >> 4, c8 = (i & 15) * 8;
    v8bf v;
    if (k < 48) v = *(const v8bf*)&V[vbase + (size_t)k * NTOT + c8];
    else {
#pragma unroll
      for (int j = 0; j < 8; ++j) v[j] = (bf16)0.f;
    }
#pragma unroll
    for (int j = 0; j < 8; ++j) Bst[c8 + j][k] = v[j];
  }
  __syncthreads();

  const int wave = tid >> 5, lane = tid & 31;
  const int lrow = lane & 15;
  const int hi   = lane >> 4;
  const int ka   = hi << 3;
  const int kbk  = hi << 3;

  v8f acc[3]; acc[0] = (v8f){}; acc[1] = (v8f){}; acc[2] = (v8f){};
#pragma unroll
  for (int kb = 0; kb < 64; kb += 32) {
    v16bf bfrag = load_b16(&Bst[wave * 16 + lrow][kb + kbk]);
#pragma unroll
    for (int mt = 0; mt < 3; ++mt) {
      v16bf a = load_a16(&As[mt * 16 + lrow][kb + ka]);
      acc[mt] = __builtin_amdgcn_wmma_f32_16x16x32_bf16(false, a, false, bfrag,
                                                        (short)0, acc[mt], false, false);
    }
  }
  // stage D (channel m, pixel n) -> Ost[pixel][channel]
  const int hi8 = hi << 3;
#pragma unroll
  for (int mt = 0; mt < 3; ++mt)
#pragma unroll
    for (int r = 0; r < 8; ++r)
      Ost[wave * 16 + lrow][mt * 16 + r + hi8] = (bf16)acc[mt][r];
  __syncthreads();
  // pixel-major global writeout, 16B chunks
  const size_t gp0 = (size_t)b * HW + n0;
  for (int i = tid; i < 128 * 6; i += 256) {
    int pl = i / 6, c8 = (i - pl * 6) * 8;
    *(v8bf*)&O[(gp0 + pl) * DIM + h * CH + c8] = *(const v8bf*)&Ost[pl][c8];
  }
}

// ---------------------------------------------------------------------------
// SK fusion: pooled mean of (fusion1 + T_D) per (b,c)
// ---------------------------------------------------------------------------
__global__ __launch_bounds__(256) void pool_kernel(const float* __restrict__ f1,
                                                   const float* __restrict__ td,
                                                   float* __restrict__ pooled) {
  const int c = blockIdx.y, b = blockIdx.x;
  const size_t fbase = (size_t)c * NTOT + (size_t)b * HW;
  const size_t tbase = ((size_t)b * DIM + c) * HW;
  float s = 0.f;
  for (int t = threadIdx.x; t < HW; t += 256) s += f1[fbase + t] + td[tbase + t];
  __shared__ float red[256];
  red[threadIdx.x] = s;
  __syncthreads();
  for (int st = 128; st > 0; st >>= 1) {
    if (threadIdx.x < st) red[threadIdx.x] += red[threadIdx.x + st];
    __syncthreads();
  }
  if (threadIdx.x == 0) pooled[b * DIM + c] = red[0] * (1.f / HW);
}

// gate MLP: z = relu(w1 @ pooled); logits = w2 @ z; softmax over HEIGHT=2
__global__ __launch_bounds__(192) void gate_kernel(const float* __restrict__ pooled,
                                                   const float* __restrict__ w1,
                                                   const float* __restrict__ w2,
                                                   float* __restrict__ gates) {
  const int b = blockIdx.x;
  __shared__ float z[24];
  const int tid = threadIdx.x;
  if (tid < 24) {
    float s = 0.f;
    for (int c = 0; c < DIM; ++c) s += w1[tid * DIM + c] * pooled[b * DIM + c];
    z[tid] = fmaxf(s, 0.f);
  }
  __syncthreads();
  float l0 = 0.f, l1 = 0.f;
#pragma unroll
  for (int j = 0; j < 24; ++j) {
    l0 += w2[tid * 24 + j] * z[j];
    l1 += w2[(DIM + tid) * 24 + j] * z[j];
  }
  float m = fmaxf(l0, l1);
  float e0 = __expf(l0 - m), e1 = __expf(l1 - m);
  gates[b * DIM + tid] = e0 / (e0 + e1);
}

// final: out = fusion1*g + T_D*(1-g), NCHW fp32 output
__global__ __launch_bounds__(256) void final_kernel(const float* __restrict__ f1,
                                                    const float* __restrict__ td,
                                                    const float* __restrict__ gates,
                                                    float* __restrict__ out) {
  size_t idx = (size_t)blockIdx.x * 256 + threadIdx.x;  // < 8*192*16384
  int hw = (int)(idx & (HW - 1));
  size_t t = idx >> 14;
  int c = (int)(t % DIM);
  int b = (int)(t / DIM);
  float g = gates[b * DIM + c];
  float f = f1[(size_t)c * NTOT + (size_t)b * HW + hw];
  out[idx] = f * g + td[idx] * (1.f - g);
}

// ---------------------------------------------------------------------------
extern "C" void kernel_launch(void* const* d_in, const int* in_sizes, int n_in,
                              void* d_out, int out_size, void* d_ws, size_t ws_size,
                              hipStream_t stream) {
  const float* T_E   = (const float*)d_in[0];
  const float* C_E   = (const float*)d_in[1];
  const float* T_D   = (const float*)d_in[2];
  const float* tn_w  = (const float*)d_in[3];
  const float* tn_b  = (const float*)d_in[4];
  const float* cn_w  = (const float*)d_in[5];
  const float* cn_b  = (const float*)d_in[6];
  const float* q_w   = (const float*)d_in[7];
  const float* q_b   = (const float*)d_in[8];
  const float* q_dw  = (const float*)d_in[9];
  const float* q_db  = (const float*)d_in[10];
  const float* k_w   = (const float*)d_in[11];
  const float* k_b   = (const float*)d_in[12];
  const float* k_dw  = (const float*)d_in[13];
  const float* k_db  = (const float*)d_in[14];
  const float* v_w   = (const float*)d_in[15];
  const float* v_b   = (const float*)d_in[16];
  const float* v_dw  = (const float*)d_in[17];
  const float* v_db  = (const float*)d_in[18];
  const float* temp  = (const float*)d_in[19];
  const float* pr_w  = (const float*)d_in[20];
  const float* pr_b  = (const float*)d_in[21];
  const float* a1    = (const float*)d_in[22];
  const float* a2    = (const float*)d_in[23];
  const float* a3    = (const float*)d_in[24];
  const float* a4    = (const float*)d_in[25];
  const float* sk_w1 = (const float*)d_in[26];
  const float* sk_w2 = (const float*)d_in[27];
  float* out = (float*)d_out;

  char* ws = (char*)d_ws;
  const size_t MB = 1ull << 20;
  bf16*  Wq     = (bf16*)(ws + 0x00000);
  bf16*  Wk     = (bf16*)(ws + 0x12000);
  bf16*  Wv     = (bf16*)(ws + 0x24000);
  bf16*  Wp     = (bf16*)(ws + 0x36000);
  float* ssq    = (float*)(ws + 0x48000);
  float* ssk    = (float*)(ws + 0x4A000);
  float* attn   = (float*)(ws + 0x4C000);   // 32*2304*4 = 288KB
  bf16*  Pbf    = (bf16*)(ws + 0x94000);    // 144KB
  float* pooled = (float*)(ws + 0xBA000);
  float* gates  = (float*)(ws + 0xBC000);
  bf16*  Xt     = (bf16*)(ws + 1 * MB);     // 48MB  [NTOT, 192] bf16
  bf16*  Xc     = (bf16*)(ws + 49 * MB);    // 48MB
  bf16*  Qbf    = (bf16*)(ws + 97 * MB);    // 48MB [192, NTOT]
  bf16*  Kbf    = (bf16*)(ws + 145 * MB);   // 48MB
  bf16*  Vbf    = (bf16*)(ws + 193 * MB);   // 48MB
  float* tmp    = (float*)(ws + 241 * MB);  // 96MB  [192, NTOT] fp32
  bf16*  Obf    = Xt;                       // alias (Xt dead after q GEMM)
  float* fusion1 = tmp;                     // alias (tmp dead after v dwconv)

  const int NW = DIM * DIM;                 // 36864
  cvt_bf16_kernel<<<(NW + 255) / 256, 256, 0, stream>>>(q_w,  Wq, NW);
  cvt_bf16_kernel<<<(NW + 255) / 256, 256, 0, stream>>>(k_w,  Wk, NW);
  cvt_bf16_kernel<<<(NW + 255) / 256, 256, 0, stream>>>(v_w,  Wv, NW);
  cvt_bf16_kernel<<<(NW + 255) / 256, 256, 0, stream>>>(pr_w, Wp, NW);

  ln_kernel<<<NTOT / 256, 256, 0, stream>>>(T_E, tn_w, tn_b, Xt);
  ln_kernel<<<NTOT / 256, 256, 0, stream>>>(C_E, cn_w, cn_b, Xc);

  dim3 ggrid(NTOT / 64, DIM / 64);          // (2048, 3)
  dim3 pgrid(NB, DIM);                      // (8, 192)

  // q path
  gemm192_kernel<<<ggrid, 256, 0, stream>>>(Wq, Xt, q_b, nullptr, tmp);
  dwconv_kernel<<<pgrid, 256, 0, stream>>>(tmp, q_dw, q_db, Qbf, ssq);
  // k path
  gemm192_kernel<<<ggrid, 256, 0, stream>>>(Wk, Xc, k_b, nullptr, tmp);
  dwconv_kernel<<<pgrid, 256, 0, stream>>>(tmp, k_dw, k_db, Kbf, ssk);
  // v path
  gemm192_kernel<<<ggrid, 256, 0, stream>>>(Wv, Xc, v_b, nullptr, tmp);
  dwconv_kernel<<<pgrid, 256, 0, stream>>>(tmp, v_dw, v_db, Vbf, nullptr);

  norm_rows_kernel<<<pgrid, 256, 0, stream>>>(Qbf, ssq);
  norm_rows_kernel<<<pgrid, 256, 0, stream>>>(Kbf, ssk);

  attn_kernel<<<NB * HEADS, 256, 0, stream>>>(Qbf, Kbf, temp, attn);
  topk_kernel<<<NB * HEADS, 64, 0, stream>>>(attn, a1, a2, a3, a4, Pbf);
  pv_kernel<<<dim3(HW / 128, NB * HEADS), 256, 0, stream>>>(Pbf, Vbf, Obf);

  // proj + residual C_E -> fusion1 (fp32, channel-major)
  gemm192_kernel<<<ggrid, 256, 0, stream>>>(Wp, Obf, pr_b, C_E, fusion1);

  pool_kernel<<<pgrid, 256, 0, stream>>>(fusion1, T_D, pooled);
  gate_kernel<<<NB, 192, 0, stream>>>(pooled, sk_w1, sk_w2, gates);
  final_kernel<<<(NB * DIM * HW) / 256, 256, 0, stream>>>(fusion1, T_D, gates, out);
}
